// WindowAttention_17377437680446
// MI455X (gfx1250) — compile-verified, matrix operand order checked by reading, not run on hardware
//
#include <hip/hip_runtime.h>
#include <hip/hip_bf16.h>

// ---------------------------------------------------------------------------
// Fused Swin window attention for gfx1250 (MI455X).
// One workgroup (8 wave32) per window; all intermediates in LDS; all GEMMs on
// v_wmma_f32_16x16x32_f16. Weights pre-converted/transposed to f16 in d_ws.
// LDS row strides padded (264/72/65) so WMMA fragment loads and the softmax
// are bank-conflict free on the 64x4B LDS banks.
// ---------------------------------------------------------------------------

typedef _Float16 half4 __attribute__((ext_vector_type(4)));
typedef _Float16 half8 __attribute__((ext_vector_type(8)));
typedef _Float16 v16h  __attribute__((ext_vector_type(16)));
typedef float    v8f   __attribute__((ext_vector_type(8)));
typedef float    f4    __attribute__((ext_vector_type(4)));

union V16U { v16h v; half8 h[2]; };

// 16x32 f16 WMMA fragment (ISA 7.12.2): lane L holds row L%16, two 8-half
// chunks at k = (L/16)*8 and k = (L/16)*8 + 16. B fragments use the same
// pattern from N-major (pre-transposed) storage.
__device__ __forceinline__ v16h load_frag16(const _Float16* __restrict__ base,
                                            int stride, int rowBase, int kBase,
                                            int lane) {
  const int row = rowBase + (lane & 15);
  const int kc  = kBase + ((lane >> 4) << 3);
  V16U u;
  u.h[0] = *reinterpret_cast<const half8*>(base + row * stride + kc);
  u.h[1] = *reinterpret_cast<const half8*>(base + row * stride + kc + 16);
  return u.v;
}

#define NTOK 49
#define CDIM 256
#define HEADS 8
#define HD 32
#define NPAD 64
#define SX 264   // f16 row stride for 256-wide matrices (528B: 16B-aligned, bank-rotating)
#define SV 72    // f16 row stride for 64-wide matrices (144B)
#define SP 72    // f16 row stride for P
#define SA 65    // f32 row stride for attn scores (conflict-free softmax)

// Convert + transpose weights to f16 N-major.  wqkvT:(768,256) wprojT:(256,256)
__global__ void swin_prep_weights(const float* __restrict__ qkvK,
                                  const float* __restrict__ projK,
                                  _Float16* __restrict__ wqkvT,
                                  _Float16* __restrict__ wprojT) {
  int idx = blockIdx.x * blockDim.x + threadIdx.x;
  if (idx < 768 * 256) {
    int j = idx >> 8, k = idx & 255;
    wqkvT[j * 256 + k] = (_Float16)qkvK[k * 768 + j];
  } else {
    int p = idx - 768 * 256;
    if (p < 256 * 256) {
      int j = p >> 8, k = p & 255;
      wprojT[j * 256 + k] = (_Float16)projK[k * 256 + j];
    }
  }
}

__global__ __launch_bounds__(256)
void swin_window_attn(const float* __restrict__ x,
                      const float* __restrict__ mask,
                      const float* __restrict__ qkv_bias,
                      const float* __restrict__ bias_table,
                      const float* __restrict__ proj_bias,
                      const int*   __restrict__ rel_index,
                      const _Float16* __restrict__ wqkvT,
                      const _Float16* __restrict__ wprojT,
                      float* __restrict__ out) {
  __shared__ alignas(16) _Float16 sQ [NPAD * SX];          // 33 KB (pre-scaled)
  __shared__ alignas(16) _Float16 sK [NPAD * SX];          // 33 KB (row-major = B^T for QK^T)
  __shared__ alignas(16) _Float16 sVt[HEADS * HD * SV];    // 36 KB (head,d,token)
  __shared__ alignas(16) _Float16 sXO[NPAD * SX];          // 33 KB: X(f16), then attn-out(f16)
  __shared__ alignas(16) float    sAttn[NTOK * SA];        // 12.4 KB
  __shared__ alignas(16) _Float16 sP [NPAD * SP];          //  9 KB
  // total ~156.5 KB

  const int tid  = threadIdx.x;
  const int lane = tid & 31;
  const int wave = __builtin_amdgcn_readfirstlane(tid >> 5);  // scalar wave id
  const int b    = blockIdx.x;
  const int wi   = b & 63;  // window-in-group for shift mask (nW = 64)

  // ---- Stage 1: load X (float4), convert f32->f16, zero-pad rows 49..63 ----
  for (int v = tid; v < NPAD * (CDIM / 4); v += 256) {
    const int r = v >> 6, c4 = (v & 63) << 2;
    f4 xv = {};
    if (r < NTOK)
      xv = *reinterpret_cast<const f4*>(x + (size_t)b * NTOK * CDIM + r * CDIM + c4);
    half4 hv = { (_Float16)xv.x, (_Float16)xv.y, (_Float16)xv.z, (_Float16)xv.w };
    *reinterpret_cast<half4*>(sXO + r * SX + c4) = hv;
  }
  for (int idx = tid; idx < (NPAD - NTOK) * NPAD; idx += 256) {
    const int r = NTOK + (idx >> 6), c = idx & 63;
    sP[r * SP + c] = (_Float16)0.f;  // pad rows of P stay zero forever
  }
  __syncthreads();

  // ---- Stage 2: QKV GEMM (64x768, K=256) -----------------------------------
  const float qscale = 0.17677669529663689f;  // 1/sqrt(32)
  for (int t = wave; t < 192; t += 8) {
    const int mT = t & 3, nT = t >> 2;     // scalar: t is wave-uniform
    const int jBase = nT * 16;
    v8f acc = {};
#pragma unroll
    for (int kk = 0; kk < 8; ++kk) {
      v16h a  = load_frag16(sXO, SX, mT * 16, kk * 32, lane);
      v16h bf = load_frag16(wqkvT, CDIM, jBase, kk * 32, lane);
      acc = __builtin_amdgcn_wmma_f32_16x16x32_f16(false, a, false, bf,
                                                   (short)0, acc, false, false);
    }
    const int j    = jBase + (lane & 15);
    const float bs = qkv_bias[j];
    const int mrow = mT * 16 + ((lane >> 4) << 3);
    if (t < 64) {                        // Q (pre-scaled)   [scalar branch]
#pragma unroll
      for (int i = 0; i < 8; ++i)
        sQ[(mrow + i) * SX + j] = (_Float16)((acc[i] + bs) * qscale);
    } else if (t < 128) {                // K
#pragma unroll
      for (int i = 0; i < 8; ++i)
        sK[(mrow + i) * SX + (j - 256)] = (_Float16)(acc[i] + bs);
    } else {                             // V, stored transposed per head
      const int jj = j - 512, h = jj >> 5, d = jj & 31;
#pragma unroll
      for (int i = 0; i < 8; ++i)
        sVt[h * (HD * SV) + d * SV + (mrow + i)] = (_Float16)(acc[i] + bs);
    }
  }
  __syncthreads();

  // ---- Stage 3: per-head attention ----------------------------------------
  const float* maskw = mask + (size_t)wi * NTOK * NTOK;
  for (int h = 0; h < HEADS; ++h) {
    // attn = Q_h @ K_h^T : hd=32 -> one k=32 WMMA per 16x16 tile
    for (int t = wave; t < 16; t += 8) {
      const int mT = t >> 2, nT = t & 3;
      v16h a  = load_frag16(sQ, SX, mT * 16, h * HD, lane);
      v16h bf = load_frag16(sK, SX, nT * 16, h * HD, lane);
      v8f acc = {};
      acc = __builtin_amdgcn_wmma_f32_16x16x32_f16(false, a, false, bf,
                                                   (short)0, acc, false, false);
      const int kc   = nT * 16 + (lane & 15);
      const int mrow = mT * 16 + ((lane >> 4) << 3);
#pragma unroll
      for (int i = 0; i < 8; ++i) {
        const int q = mrow + i;
        if (q < NTOK) {
          float val = -1e30f;  // pad cols -> exp()==0
          if (kc < NTOK) {
            const int ri = rel_index[q * NTOK + kc];
            val = acc[i] + bias_table[ri * HEADS + h] + maskw[q * NTOK + kc];
          }
          sAttn[q * SA + kc] = val;
        }
      }
    }
    __syncthreads();

    // row softmax (49 rows; stride 65 -> conflict-free)
    if (tid < NTOK) {
      const float* rp = sAttn + tid * SA;
      float mx = -1e30f;
      for (int k2 = 0; k2 < NTOK; ++k2) mx = fmaxf(mx, rp[k2]);
      float s = 0.f;
      for (int k2 = 0; k2 < NTOK; ++k2) s += __expf(rp[k2] - mx);
      const float inv = 1.0f / s;
      for (int k2 = 0; k2 < NTOK; ++k2)
        sP[tid * SP + k2] = (_Float16)(__expf(rp[k2] - mx) * inv);
      for (int k2 = NTOK; k2 < NPAD; ++k2) sP[tid * SP + k2] = (_Float16)0.f;
    }
    __syncthreads();

    // out_h = P @ V_h : 8 tiles (4M x 2N), one per wave, K=64 -> 2 steps
    {
      const int mT = wave >> 1, nT = wave & 1;
      v8f acc = {};
#pragma unroll
      for (int ks = 0; ks < 2; ++ks) {
        v16h a  = load_frag16(sP, SP, mT * 16, ks * 32, lane);
        v16h bf = load_frag16(sVt + h * (HD * SV), SV, nT * 16, ks * 32, lane);
        acc = __builtin_amdgcn_wmma_f32_16x16x32_f16(false, a, false, bf,
                                                     (short)0, acc, false, false);
      }
      const int jcol = h * HD + nT * 16 + (lane & 15);
      const int mrow = mT * 16 + ((lane >> 4) << 3);
#pragma unroll
      for (int i = 0; i < 8; ++i)
        sXO[(mrow + i) * SX + jcol] = (_Float16)acc[i];
    }
    __syncthreads();
  }

  // ---- Stage 4: output projection (64x256, K=256) + bias -> HBM -----------
  for (int t = wave; t < 64; t += 8) {
    const int mT = t & 3, nT = t >> 2;
    v8f acc = {};
#pragma unroll
    for (int kk = 0; kk < 8; ++kk) {
      v16h a  = load_frag16(sXO, SX, mT * 16, kk * 32, lane);
      v16h bf = load_frag16(wprojT, CDIM, nT * 16, kk * 32, lane);
      acc = __builtin_amdgcn_wmma_f32_16x16x32_f16(false, a, false, bf,
                                                   (short)0, acc, false, false);
    }
    const int jcol = nT * 16 + (lane & 15);
    const int mrow = mT * 16 + ((lane >> 4) << 3);
    const float pb = proj_bias[jcol];
#pragma unroll
    for (int i = 0; i < 8; ++i) {
      const int q = mrow + i;
      if (q < NTOK)
        out[(size_t)b * NTOK * CDIM + q * CDIM + jcol] = acc[i] + pb;
    }
  }
}

extern "C" void kernel_launch(void* const* d_in, const int* in_sizes, int n_in,
                              void* d_out, int out_size, void* d_ws, size_t ws_size,
                              hipStream_t stream) {
  const float* x         = (const float*)d_in[0];
  const float* mask      = (const float*)d_in[1];
  const float* qkvK      = (const float*)d_in[2];
  const float* qkv_bias  = (const float*)d_in[3];
  const float* bias_tab  = (const float*)d_in[4];
  const float* projK     = (const float*)d_in[5];
  const float* proj_bias = (const float*)d_in[6];
  const int*   rel_index = (const int*)d_in[7];

  _Float16* wqkvT  = (_Float16*)d_ws;          // 768*256 f16 = 384 KB
  _Float16* wprojT = wqkvT + 768 * 256;        // 256*256 f16 = 128 KB

  swin_prep_weights<<<1024, 256, 0, stream>>>(qkvK, projK, wqkvT, wprojT);
  swin_window_attn<<<4096, 256, 0, stream>>>(x, mask, qkv_bias, bias_tab,
                                             proj_bias, rel_index,
                                             wqkvT, wprojT, (float*)d_out);
}
